// VaeDecoder_49512382988527
// MI455X (gfx1250) — compile-verified, matrix operand order checked by reading, not run on hardware
//
#include <hip/hip_runtime.h>
#include <hip/hip_bf16.h>

// ---------------------------------------------------------------------------
// VAE LSTM decoder for MI455X (gfx1250, wave32, WMMA).
// Batch rows are independent recurrent chains -> 16 persistent workgroups,
// each owns a 16-row batch tile (M=16 == WMMA tile). bf16 weights streamed
// from L2 (whole model ~10 MB bf16 << 192 MB L2); h ping-pong + x||cond in
// LDS; c held in registers as WMMA D-fragments.
// ---------------------------------------------------------------------------

typedef __attribute__((ext_vector_type(8)))  __bf16 v8bf;
typedef __attribute__((ext_vector_type(16))) __bf16 v16bf;
typedef __attribute__((ext_vector_type(8)))  float  v8f;

#define LATENT 256
#define CONDD  64
#define HID    1024
#define OUTD   128
#define TLEN   512
#define BT     16              // batch tile per workgroup
#define HP     (HID + 8)       // padded h pitch (keeps 16B alignment: 1032*2 % 16 == 0)
#define XK     (OUTD + CONDD)  // 192
#define XP     (XK + 8)        // padded x pitch (200*2 % 16 == 0)

__device__ __forceinline__ float sigm(float x) { return 1.0f / (1.0f + __expf(-x)); }
__device__ __forceinline__ float tanh_fast(float x) {
    float e = __expf(2.0f * x);
    return (e - 1.0f) / (e + 1.0f);
}

// One 16x16x32 bf16 operand fragment for this lane.
// Per ISA layout: half = lane>>4 selects k-offset {0,+8}; the 16 values are
// two contiguous 8-element (16B) chunks at k and k+16 -> two b128 loads.
__device__ __forceinline__ v16bf load_frag(const __bf16* p) {
    v8bf lo = *(const v8bf*)(p);
    v8bf hi = *(const v8bf*)(p + 16);
    return __builtin_shufflevector(lo, hi, 0,1,2,3,4,5,6,7,8,9,10,11,12,13,14,15);
}

__device__ __forceinline__ v8f wmma_bf16(v16bf a, v16bf b, v8f c) {
    return __builtin_amdgcn_wmma_f32_16x16x32_bf16(false, a, false, b, (short)0, c,
                                                   false, false);
}

// ---------------------------------------------------------------------------
// Prologue: fp32 -> bf16 weight conversion + fused LSTM bias.
// ---------------------------------------------------------------------------
extern "C" __global__ void vae_convert_kernel(
    const float* __restrict__ Wih_f, const float* __restrict__ Whh_f,
    const float* __restrict__ Wout_f,
    const float* __restrict__ b_ih, const float* __restrict__ b_hh,
    __bf16* __restrict__ Wih, __bf16* __restrict__ Whh, __bf16* __restrict__ Wout,
    float* __restrict__ biasg)
{
    size_t i = (size_t)blockIdx.x * blockDim.x + threadIdx.x;
    size_t stride = (size_t)gridDim.x * blockDim.x;
    for (size_t k = i; k < (size_t)4 * HID * XK;  k += stride) Wih[k]  = (__bf16)Wih_f[k];
    for (size_t k = i; k < (size_t)4 * HID * HID; k += stride) Whh[k]  = (__bf16)Whh_f[k];
    for (size_t k = i; k < (size_t)OUTD * HID;    k += stride) Wout[k] = (__bf16)Wout_f[k];
    for (size_t k = i; k < (size_t)4 * HID;       k += stride) biasg[k] = b_ih[k] + b_hh[k];
}

// ---------------------------------------------------------------------------
// Persistent LSTM kernel: 16 blocks x 256 threads (8 waves).
// Wave w owns hidden columns [w*128, w*128+128).
// ---------------------------------------------------------------------------
extern "C" __global__ void __launch_bounds__(256, 1)
vae_lstm_main(const float* __restrict__ latent, const float* __restrict__ cond,
              const int* __restrict__ lens,
              const float* __restrict__ W_h, const float* __restrict__ b_h,
              const float* __restrict__ W_c, const float* __restrict__ b_c,
              const float* __restrict__ W_s, const float* __restrict__ b_s,
              const __bf16* __restrict__ Wih, const __bf16* __restrict__ Whh,
              const __bf16* __restrict__ Wout,
              const float* __restrict__ biasg, const float* __restrict__ b_out,
              float* __restrict__ out)
{
    __shared__ __align__(16) __bf16 hbuf[2][BT][HP];   // ping-pong hidden state (bf16)
    __shared__ __align__(16) __bf16 xcat[BT][XP];      // [y | conditions] (bf16)
    __shared__ float sbias[4 * HID];
    __shared__ float sbout[OUTD];
    __shared__ int   slens[BT];

    const int tid  = threadIdx.x;
    const int wave = tid >> 5;
    const int lane = tid & 31;
    const int col  = lane & 15;    // M for A-frags / N for B,D-frags
    const int half = lane >> 4;
    const int b0   = blockIdx.x * BT;

    // ---- init: biases / lengths into LDS ----
    for (int i = tid; i < 4 * HID; i += 256) sbias[i] = biasg[i];
    for (int i = tid; i < OUTD;    i += 256) sbout[i] = b_out[i];
    if (tid < BT) slens[tid] = lens[b0 + tid];

    // ---- x0 = latent @ W_s^T + b_s  -> xcat[:, 0:128] ----
    for (int idx = tid; idx < BT * OUTD; idx += 256) {
        int m = idx >> 7, o = idx & (OUTD - 1);
        const float* lv = latent + (size_t)(b0 + m) * LATENT;
        const float* wr = W_s + (size_t)o * LATENT;
        float acc = b_s[o];
        for (int q = 0; q < LATENT; ++q) acc = __builtin_fmaf(lv[q], wr[q], acc);
        xcat[m][o] = (__bf16)acc;
    }
    // ---- conditions -> xcat[:, 128:192] ----
    for (int idx = tid; idx < BT * CONDD; idx += 256) {
        int m = idx >> 6, c = idx & (CONDD - 1);
        xcat[m][OUTD + c] = (__bf16)cond[(size_t)(b0 + m) * CONDD + c];
    }
    // ---- h0 = latent @ W_h^T + b_h -> hbuf[0] (bf16) ----
    for (int idx = tid; idx < BT * HID; idx += 256) {
        int m = idx >> 10, k = idx & (HID - 1);
        const float* lv = latent + (size_t)(b0 + m) * LATENT;
        const float* wr = W_h + (size_t)k * LATENT;
        float acc = b_h[k];
        for (int q = 0; q < LATENT; ++q) acc = __builtin_fmaf(lv[q], wr[q], acc);
        hbuf[0][m][k] = (__bf16)acc;
    }
    // ---- c0 in D-fragment layout: element v <-> row (v + 8*half), col (lane&15) ----
    v8f cfrag[8];
    #pragma unroll
    for (int j = 0; j < 8; ++j) {
        const int n = wave * 128 + j * 16 + col;
        const float* wc = W_c + (size_t)n * LATENT;
        const float bcv = b_c[n];
        v8f cf = {};
        #pragma unroll 1
        for (int v = 0; v < 8; ++v) {
            const float* lv = latent + (size_t)(b0 + v + 8 * half) * LATENT;
            float acc = bcv;
            for (int q = 0; q < LATENT; ++q) acc = __builtin_fmaf(lv[q], wc[q], acc);
            cf[v] = acc;
        }
        cfrag[j] = cf;
    }
    __syncthreads();

    // ---- 512-step recurrence ----
    for (int t = 0; t < TLEN; ++t) {
        const __bf16 (*hc)[HP] = hbuf[t & 1];        // read
        __bf16 (*hn)[HP]       = hbuf[(t & 1) ^ 1];  // write

        // Phase 1: gates for this wave's 128 columns, all four gates.
        #pragma unroll
        for (int j = 0; j < 8; ++j) {
            const int n0 = wave * 128 + j * 16;
            const int nc = n0 + col;
            v8f acc[4] = {};

            // x||cond part: K = 192 (6 k-tiles), B rows = W_ih[g*1024+nc][.]
            #pragma unroll 1
            for (int kt = 0; kt < XK / 32; ++kt) {
                v16bf af = load_frag(&xcat[col][kt * 32 + half * 8]);
                #pragma unroll
                for (int g = 0; g < 4; ++g) {
                    const __bf16* bp =
                        Wih + (size_t)(g * HID + nc) * XK + kt * 32 + half * 8;
                    acc[g] = wmma_bf16(af, load_frag(bp), acc[g]);
                }
            }
            // h part: K = 1024 (32 k-tiles), B rows = W_hh[g*1024+nc][.]
            #pragma unroll 2
            for (int kt = 0; kt < HID / 32; ++kt) {
                v16bf af = load_frag(&hc[col][kt * 32 + half * 8]);
                if (kt + 1 < HID / 32)
                    __builtin_prefetch(Whh + (size_t)nc * HID + (kt + 1) * 32, 0, 0);
                #pragma unroll
                for (int g = 0; g < 4; ++g) {
                    const __bf16* bp =
                        Whh + (size_t)(g * HID + nc) * HID + kt * 32 + half * 8;
                    acc[g] = wmma_bf16(af, load_frag(bp), acc[g]);
                }
            }

            // LSTM elementwise on this wave's columns; c stays in registers.
            const float bi  = sbias[nc];
            const float bfv = sbias[HID + nc];
            const float bg  = sbias[2 * HID + nc];
            const float bo  = sbias[3 * HID + nc];
            v8f cf = cfrag[j];
            #pragma unroll
            for (int v = 0; v < 8; ++v) {
                float iv = sigm(acc[0][v] + bi);
                float fv = sigm(acc[1][v] + bfv);
                float gv = tanh_fast(acc[2][v] + bg);
                float ov = sigm(acc[3][v] + bo);
                float cn = fv * cf[v] + iv * gv;
                cf[v] = cn;
                hn[v + 8 * half][nc] = (__bf16)(ov * tanh_fast(cn));
            }
            cfrag[j] = cf;
        }
        __syncthreads();   // hn complete for everyone

        // Phase 2: y = h_new @ W_out^T + b_out; wave w -> output cols [w*16, w*16+16)
        {
            const int n0 = wave * 16;
            v8f acc = {};
            #pragma unroll 2
            for (int kt = 0; kt < HID / 32; ++kt) {
                v16bf af = load_frag(&hn[col][kt * 32 + half * 8]);
                const __bf16* bp =
                    Wout + (size_t)(n0 + col) * HID + kt * 32 + half * 8;
                acc = wmma_bf16(af, load_frag(bp), acc);
            }
            const float bov = sbout[n0 + col];
            #pragma unroll
            for (int v = 0; v < 8; ++v) {
                const int m = v + 8 * half;
                float y = acc[v] + bov;
                out[(((size_t)(b0 + m)) * TLEN + t) * OUTD + n0 + col] =
                    (t < slens[m]) ? y : 0.0f;
                xcat[m][n0 + col] = (__bf16)y;   // next step's x (unmasked, per reference)
            }
        }
        __syncthreads();   // xcat ready for next step
    }
}

// ---------------------------------------------------------------------------
// Host launcher
// ---------------------------------------------------------------------------
extern "C" void kernel_launch(void* const* d_in, const int* in_sizes, int n_in,
                              void* d_out, int out_size, void* d_ws, size_t ws_size,
                              hipStream_t stream) {
    (void)in_sizes; (void)n_in; (void)out_size; (void)ws_size;

    const float* latent = (const float*)d_in[0];
    const float* cond   = (const float*)d_in[1];
    const int*   lens   = (const int*)d_in[2];
    const float* W_h    = (const float*)d_in[3];
    const float* b_h    = (const float*)d_in[4];
    const float* W_c    = (const float*)d_in[5];
    const float* b_c    = (const float*)d_in[6];
    const float* W_s    = (const float*)d_in[7];
    const float* b_s    = (const float*)d_in[8];
    const float* W_ih   = (const float*)d_in[9];
    const float* W_hh   = (const float*)d_in[10];
    const float* b_ih   = (const float*)d_in[11];
    const float* b_hh   = (const float*)d_in[12];
    const float* W_out  = (const float*)d_in[13];
    const float* b_out  = (const float*)d_in[14];

    // Workspace carve-out (bf16 weights + fused bias), 256B aligned slots.
    char* ws = (char*)d_ws;
    size_t off = 0;
    auto carve = [&](size_t bytes) -> char* {
        char* p = ws + off;
        off = (off + bytes + 255) & ~(size_t)255;
        return p;
    };
    __bf16* Wih_b  = (__bf16*)carve((size_t)4 * HID * XK * sizeof(__bf16));   // 1.5 MB
    __bf16* Whh_b  = (__bf16*)carve((size_t)4 * HID * HID * sizeof(__bf16));  // 8.0 MB
    __bf16* Wout_b = (__bf16*)carve((size_t)OUTD * HID * sizeof(__bf16));     // 0.25 MB
    float*  biasg  = (float*)carve((size_t)4 * HID * sizeof(float));          // 16 KB

    vae_convert_kernel<<<512, 256, 0, stream>>>(W_ih, W_hh, W_out, b_ih, b_hh,
                                                Wih_b, Whh_b, Wout_b, biasg);

    vae_lstm_main<<<256 / BT, 256, 0, stream>>>(latent, cond, lens,
                                                W_h, b_h, W_c, b_c, W_s, b_s,
                                                Wih_b, Whh_b, Wout_b,
                                                biasg, b_out, (float*)d_out);
}
